// Block_77146202571190
// MI455X (gfx1250) — matrix operand order repaired and checked by reading, not compiled, not run on hardware
//
#include <hip/hip_runtime.h>
#include <cstdint>

// ---------------------------------------------------------------------------
// Problem constants (from reference)
// ---------------------------------------------------------------------------
#define NTOK 65536
#define CDIM 512
#define NHEAD 8
#define DHEAD 64
#define KWIN 128
#define LRANK 16
#define HIDD 2048
#define NWIN (NTOK / KWIN)          // 512
#define LORA_SCALE 2.0f
#define ATTN_SCALE 0.125f           // 64^-0.5

typedef __bf16 bf16_t;
typedef __attribute__((ext_vector_type(16))) __bf16 v16bf;
typedef __attribute__((ext_vector_type(8)))  float  v8f;

// ---------------------------------------------------------------------------
// Optional CDNA5 async global->LDS path (ASYNCcnt-tracked LDS DMA).
// Compile-safe: only enabled if the clang builtins exist on this toolchain.
// Builtin signature (from hipcc diagnostic): void(int4 AS1*, int4 AS3*, Ii, Ii)
// ---------------------------------------------------------------------------
#if defined(__has_builtin)
#if __has_builtin(__builtin_amdgcn_global_load_async_to_lds_b128) && \
    __has_builtin(__builtin_amdgcn_s_wait_asynccnt)
#define HAVE_ASYNC_LDS 1
#endif
#endif

#if HAVE_ASYNC_LDS
typedef int v4i_gcc __attribute__((vector_size(16)));
typedef __attribute__((address_space(1))) v4i_gcc as1_v4i;
typedef __attribute__((address_space(3))) v4i_gcc as3_v4i;
// Generic LDS pointers carry the LDS byte offset in their low 32 bits
// (flat-aperture rule, ISA 10.2), so the integer round-trip yields a valid
// AS(3) pointer; global generic pointers are already the 64-bit VA.
__device__ __forceinline__ void async_copy_b128(const bf16_t* gsrc, bf16_t* ldst) {
    as1_v4i* g = (as1_v4i*)(unsigned long long)(uintptr_t)gsrc;
    as3_v4i* l = (as3_v4i*)(unsigned int)(uintptr_t)ldst;
    __builtin_amdgcn_global_load_async_to_lds_b128(g, l, 0, 0);
}
__device__ __forceinline__ void async_wait0() {
    __builtin_amdgcn_s_wait_asynccnt(0);
}
#else
__device__ __forceinline__ void async_wait0() {}
#endif

// ---------------------------------------------------------------------------
// Fragment helpers (wave32 WMMA, 16-bit operands; contiguous k order used by this file)
//  A 16x32: lane<16 -> row=lane, K = 0..15 contiguous; lane>=16 -> K=16..31
//  B 32x16: lane<16 -> col=lane, K = 0..15 contiguous; lane>=16 -> K=16..31
// ---------------------------------------------------------------------------
__device__ inline v16bf load_frag_a(const bf16_t* base, int kb) {
    v16bf a;
#pragma unroll
    for (int i = 0; i < 8; ++i) a[i] = base[kb + i];
#pragma unroll
    for (int i = 0; i < 8; ++i) a[8 + i] = base[kb + 8 + i];
    return a;
}

__device__ inline v16bf load_frag16(const bf16_t* p) {
    v16bf b;
#pragma unroll
    for (int i = 0; i < 16; ++i) b[i] = p[i];
    return b;
}

__device__ inline v8f wmma_bf16(v16bf a, v16bf b, v8f c) {
    return __builtin_amdgcn_wmma_f32_16x16x32_bf16(
        false, a, false, b, (short)0, c, false, false);
}

// ---------------------------------------------------------------------------
// Elementwise kernels
// ---------------------------------------------------------------------------
__global__ void cast_f32_bf16_kernel(const float* __restrict__ x,
                                     bf16_t* __restrict__ y, long long n) {
    long long i = (long long)blockIdx.x * blockDim.x + threadIdx.x;
    long long st = (long long)gridDim.x * blockDim.x;
    for (; i < n; i += st) y[i] = (bf16_t)x[i];
}

__global__ void gelu_cast_kernel(const float* __restrict__ x,
                                 bf16_t* __restrict__ y, long long n) {
    long long i = (long long)blockIdx.x * blockDim.x + threadIdx.x;
    long long st = (long long)gridDim.x * blockDim.x;
    for (; i < n; i += st) {
        float v = x[i];
        float c = 0.7978845608f * (v + 0.044715f * v * v * v);
        y[i] = (bf16_t)(0.5f * v * (1.0f + tanhf(c)));
    }
}

// LayerNorm over C=512, one block (256 threads) per row, bf16 output
__global__ __launch_bounds__(256)
void ln_cast_kernel(const float* __restrict__ x, const float* __restrict__ g,
                    const float* __restrict__ bta, bf16_t* __restrict__ y) {
    int row = blockIdx.x;
    const float* xr = x + (size_t)row * CDIM;
    int tid = threadIdx.x;
    float v0 = xr[tid], v1 = xr[tid + 256];
    float s = v0 + v1, sq = v0 * v0 + v1 * v1;
#pragma unroll
    for (int m = 1; m < 32; m <<= 1) {
        s  += __shfl_xor(s,  m, 32);
        sq += __shfl_xor(sq, m, 32);
    }
    __shared__ float rs[8], rq[8], mb[2];
    int lane = tid & 31, wv = tid >> 5;
    if (lane == 0) { rs[wv] = s; rq[wv] = sq; }
    __syncthreads();
    if (tid == 0) {
        float S = 0.f, Q = 0.f;
#pragma unroll
        for (int i = 0; i < 8; ++i) { S += rs[i]; Q += rq[i]; }
        float mean = S * (1.0f / CDIM);
        float var  = Q * (1.0f / CDIM) - mean * mean;
        mb[0] = mean; mb[1] = rsqrtf(var + 1e-5f);
    }
    __syncthreads();
    float mean = mb[0], inv = mb[1];
    bf16_t* yr = y + (size_t)row * CDIM;
    yr[tid]       = (bf16_t)((v0 - mean) * inv * g[tid]       + bta[tid]);
    yr[tid + 256] = (bf16_t)((v1 - mean) * inv * g[tid + 256] + bta[tid + 256]);
}

// ---------------------------------------------------------------------------
// Tiled bf16 WMMA GEMM:  C[M x Nc] (+)= alpha * A[M x K] @ B[K x Nc]
//   accum==0: C = alpha*AB + bias + res (bias/res nullable)
//   accum==1: C += alpha*AB
//   Cb (nullable): bf16 mirror of final C
// Block tile 128x128, K-step 32, 8 waves; wave w owns rows [16w,16w+16).
// Double-buffered LDS; batched fragment loads -> 8 back-to-back WMMAs.
// ---------------------------------------------------------------------------
#define BM 128
#define BN 128
#define BK 32
#define LDT 40   // LDS leading dim (elems); 80B rows -> 16B aligned fragments

__global__ __launch_bounds__(256)
void gemm_bf16_kernel(const bf16_t* __restrict__ A, const bf16_t* __restrict__ B,
                      const float* __restrict__ bias, const float* __restrict__ res,
                      float* __restrict__ C, bf16_t* __restrict__ Cb,
                      int M, int Ncol, int Kd, float alpha, int accum) {
    __shared__ bf16_t As[2][BM * LDT];
    __shared__ bf16_t Bs[2][BN * LDT];   // B tile stored transposed: [n][k]

    const int m0 = blockIdx.y * BM;
    const int n0 = blockIdx.x * BN;
    const int tid = threadIdx.x;
    const int lane = tid & 31, wv = tid >> 5;
    const int hf = lane >> 4, l16 = lane & 15;

    v8f zero = {};
    v8f acc[8];
#pragma unroll
    for (int t = 0; t < 8; ++t) acc[t] = zero;

    const int ar = tid >> 1;            // A stage row 0..127
    const int ac = (tid & 1) * 16;      // A stage col 0 or 16

    auto stageA = [&](int buf, int k0) {
        const bf16_t* src = A + (size_t)(m0 + ar) * Kd + k0 + ac;
        bf16_t* dst = &As[buf][ar * LDT + ac];
        if (k0 + BK <= Kd) {
#if HAVE_ASYNC_LDS
            async_copy_b128(src, dst);
            async_copy_b128(src + 8, dst + 8);
#else
            float4 u0 = ((const float4*)src)[0];
            float4 u1 = ((const float4*)src)[1];
            ((float4*)dst)[0] = u0;
            ((float4*)dst)[1] = u1;
#endif
            if (k0 + BK < Kd) __builtin_prefetch(src + BK, 0, 1);  // global_prefetch_b8
        } else {
#pragma unroll
            for (int i = 0; i < 16; ++i)
                dst[i] = (k0 + ac + i < Kd) ? src[i] : (bf16_t)0.f;
        }
    };
    auto stageB = [&](int buf, int k0) {
#pragma unroll
        for (int rep = 0; rep < 2; ++rep) {
            int idx = tid + 256 * rep;
            int kr = idx >> 4;           // 0..31
            int cg = (idx & 15) * 8;     // 0..120
            bool kok = (k0 + kr) < Kd;
            if (kok && (n0 + cg + 8) <= Ncol) {
                float4 u = *((const float4*)(B + (size_t)(k0 + kr) * Ncol + n0 + cg));
                const bf16_t* tv = (const bf16_t*)&u;
#pragma unroll
                for (int i = 0; i < 8; ++i) Bs[buf][(cg + i) * LDT + kr] = tv[i];
            } else {
#pragma unroll
                for (int i = 0; i < 8; ++i) {
                    bf16_t v = (bf16_t)0.f;
                    if (kok && (n0 + cg + i) < Ncol)
                        v = B[(size_t)(k0 + kr) * Ncol + n0 + cg + i];
                    Bs[buf][(cg + i) * LDT + kr] = v;
                }
            }
        }
    };

    const int nk = (Kd + BK - 1) / BK;
    stageA(0, 0);
    stageB(0, 0);
    async_wait0();
    __syncthreads();

    for (int ik = 0; ik < nk; ++ik) {
        const int buf = ik & 1;
        if (ik + 1 < nk) {               // issue next tile before compute
            stageA(buf ^ 1, (ik + 1) * BK);
            stageB(buf ^ 1, (ik + 1) * BK);
        }
        // ---- compute current tile: batch all fragment loads, then 8 WMMAs ----
        const bf16_t* abase = &As[buf][(wv * 16 + l16) * LDT];
        v16bf af = load_frag_a(abase, 16 * hf);
        v16bf bfr[8];
#pragma unroll
        for (int nt = 0; nt < 8; ++nt)
            bfr[nt] = load_frag16(&Bs[buf][(nt * 16 + l16) * LDT + 16 * hf]);
#pragma unroll
        for (int nt = 0; nt < 8; ++nt)
            acc[nt] = wmma_bf16(af, bfr[nt], acc[nt]);
        async_wait0();
        __syncthreads();
    }

    // ---- epilogue (C layout: VGPR r, lanes<16 -> M=r, lanes>=16 -> M=r+8) ----
#pragma unroll
    for (int nt = 0; nt < 8; ++nt) {
        int col = n0 + nt * 16 + l16;
        if (col >= Ncol) continue;
#pragma unroll
        for (int r = 0; r < 8; ++r) {
            int row = m0 + wv * 16 + r + 8 * hf;
            if (row >= M) continue;
            size_t idx = (size_t)row * Ncol + col;
            float v = acc[nt][r] * alpha;
            if (accum) {
                float o = C[idx] + v;
                C[idx] = o;
                if (Cb) Cb[idx] = (bf16_t)o;
            } else {
                if (bias) v += bias[col];
                if (res)  v += res[idx];
                C[idx] = v;
                if (Cb) Cb[idx] = (bf16_t)v;
            }
        }
    }
}

// ---------------------------------------------------------------------------
// Windowed attention, one block per (window, head).
// qkvb: [N][3][H][D] bf16 (GEMM output incl. LoRA). Gathered via order.
// obf : [N][C] bf16, scattered back via order (== [inverse] in reference).
// ---------------------------------------------------------------------------
#define QK_LD 72    // 144B rows -> 16B aligned
#define VT_LD 136   // 272B rows -> 16B aligned
#define SMEM_ELEMS (128 * QK_LD * 2 + 64 * VT_LD)   // 27136 elems = 54272 B

__global__ __launch_bounds__(256)
void attn_kernel(const bf16_t* __restrict__ qkvb, const int* __restrict__ order,
                 bf16_t* __restrict__ obf) {
    __shared__ bf16_t smem[SMEM_ELEMS];
    bf16_t* qs = smem;                    // [128][QK_LD]
    bf16_t* ks = qs + 128 * QK_LD;        // [128][QK_LD]
    bf16_t* vT = ks + 128 * QK_LD;        // [64][VT_LD]   (v transposed)
    bf16_t* ps = smem;                    // overlay on qs+ks after S phase:
                                          // per-wave [16][VT_LD] (8*2176=17408 el)
    const int w = blockIdx.x, hd = blockIdx.y;
    const int tid = threadIdx.x;
    const int lane = tid & 31, wv = tid >> 5;
    const int hf = lane >> 4, l16 = lane & 15;

    // ---- gather q,k,v for this (window, head) into LDS ----
    {
        int row = tid >> 1;
        int off = (tid & 1) * 32;
        int g = order[w * KWIN + row];
        const bf16_t* src = qkvb + (size_t)g * (3 * CDIM) + hd * DHEAD;
#if HAVE_ASYNC_LDS
        // q,k: direct copies via async LDS DMA (ASYNCcnt)
        async_copy_b128(src + off,              qs + row * QK_LD + off);
        async_copy_b128(src + off + 8,          qs + row * QK_LD + off + 8);
        async_copy_b128(src + off + 16,         qs + row * QK_LD + off + 16);
        async_copy_b128(src + off + 24,         qs + row * QK_LD + off + 24);
        async_copy_b128(src + CDIM + off,       ks + row * QK_LD + off);
        async_copy_b128(src + CDIM + off + 8,   ks + row * QK_LD + off + 8);
        async_copy_b128(src + CDIM + off + 16,  ks + row * QK_LD + off + 16);
        async_copy_b128(src + CDIM + off + 24,  ks + row * QK_LD + off + 24);
#pragma unroll
        for (int c = 0; c < 32; c += 8) {
            float4 uv = *(const float4*)(src + 2 * CDIM + off + c);
            const bf16_t* tv = (const bf16_t*)&uv;
#pragma unroll
            for (int i = 0; i < 8; ++i)
                vT[(off + c + i) * VT_LD + row] = tv[i];
        }
#else
#pragma unroll
        for (int c = 0; c < 32; c += 8) {
            float4 uq = *(const float4*)(src + off + c);
            float4 uk = *(const float4*)(src + CDIM + off + c);
            float4 uv = *(const float4*)(src + 2 * CDIM + off + c);
            *(float4*)(qs + row * QK_LD + off + c) = uq;
            *(float4*)(ks + row * QK_LD + off + c) = uk;
            const bf16_t* tv = (const bf16_t*)&uv;
#pragma unroll
            for (int i = 0; i < 8; ++i)
                vT[(off + c + i) * VT_LD + row] = tv[i];
        }
#endif
    }
    async_wait0();
    __syncthreads();

    // ---- S = scale * q @ k^T, wave owns 16 rows x 128 cols ----
    v8f zero = {};
    v8f sacc[8];
#pragma unroll
    for (int t = 0; t < 8; ++t) sacc[t] = zero;

    const bf16_t* abase = qs + (wv * 16 + l16) * QK_LD;
#pragma unroll
    for (int kk = 0; kk < DHEAD; kk += 32) {
        v16bf af = load_frag_a(abase + kk, 16 * hf);
        v16bf bk[8];
#pragma unroll
        for (int nt = 0; nt < 8; ++nt)
            bk[nt] = load_frag16(ks + (nt * 16 + l16) * QK_LD + kk + 16 * hf);
#pragma unroll
        for (int nt = 0; nt < 8; ++nt)
            sacc[nt] = wmma_bf16(af, bk[nt], sacc[nt]);
    }

    // ---- softmax in registers (row M=r+8*hf lives in 16-lane half-waves) ----
#pragma unroll
    for (int r = 0; r < 8; ++r) {
        float mx = -3.4e38f;
#pragma unroll
        for (int nt = 0; nt < 8; ++nt) {
            float v = sacc[nt][r] * ATTN_SCALE;
            sacc[nt][r] = v;
            mx = fmaxf(mx, v);
        }
#pragma unroll
        for (int d = 1; d < 16; d <<= 1) mx = fmaxf(mx, __shfl_xor(mx, d, 32));
        float sm = 0.f;
#pragma unroll
        for (int nt = 0; nt < 8; ++nt) {
            float e = __expf(sacc[nt][r] - mx);
            sacc[nt][r] = e;
            sm += e;
        }
#pragma unroll
        for (int d = 1; d < 16; d <<= 1) sm += __shfl_xor(sm, d, 32);
        float inv = 1.f / sm;
#pragma unroll
        for (int nt = 0; nt < 8; ++nt) sacc[nt][r] *= inv;
    }

    // ---- P -> bf16 in per-wave LDS region (overlay; q/k dead now) ----
    __syncthreads();
    bf16_t* pw = ps + wv * 16 * VT_LD;
#pragma unroll
    for (int nt = 0; nt < 8; ++nt)
#pragma unroll
        for (int r = 0; r < 8; ++r)
            pw[(r + 8 * hf) * VT_LD + nt * 16 + l16] = (bf16_t)sacc[nt][r];

    // ---- O = P @ V : 16x64 per wave, K=128 ----
    v8f oacc[4];
#pragma unroll
    for (int t = 0; t < 4; ++t) oacc[t] = zero;
    const bf16_t* pbase = pw + l16 * VT_LD;
#pragma unroll
    for (int kk = 0; kk < KWIN; kk += 32) {
        v16bf af = load_frag_a(pbase + kk, 16 * hf);
        v16bf bv[4];
#pragma unroll
        for (int ot = 0; ot < 4; ++ot)
            bv[ot] = load_frag16(vT + (ot * 16 + l16) * VT_LD + kk + 16 * hf);
#pragma unroll
        for (int ot = 0; ot < 4; ++ot)
            oacc[ot] = wmma_bf16(af, bv[ot], oacc[ot]);
    }

    // ---- scatter back to original token order (== [inverse]) ----
    int grows[8];
#pragma unroll
    for (int r = 0; r < 8; ++r)
        grows[r] = order[w * KWIN + wv * 16 + r + 8 * hf];
#pragma unroll
    for (int ot = 0; ot < 4; ++ot) {
        int col = ot * 16 + l16;
#pragma unroll
        for (int r = 0; r < 8; ++r)
            obf[(size_t)grows[r] * CDIM + hd * DHEAD + col] = (bf16_t)oacc[ot][r];
    }
}

// ---------------------------------------------------------------------------
// Host-side orchestration
// ---------------------------------------------------------------------------
extern "C" void kernel_launch(void* const* d_in, const int* in_sizes, int n_in,
                              void* d_out, int out_size, void* d_ws, size_t ws_size,
                              hipStream_t stream) {
    (void)in_sizes; (void)n_in; (void)out_size; (void)ws_size;
    const float* feat  = (const float*)d_in[0];
    const int*   order = (const int*)d_in[1];
    // d_in[2] = inverse (unused: scatter via order is equivalent)
    const float* g1 = (const float*)d_in[3];  const float* b1   = (const float*)d_in[4];
    const float* Wqkv = (const float*)d_in[5]; const float* bqkv = (const float*)d_in[6];
    const float* Aqkv = (const float*)d_in[7]; const float* Bqkv = (const float*)d_in[8];
    const float* Wproj = (const float*)d_in[9]; const float* bproj = (const float*)d_in[10];
    const float* Aproj = (const float*)d_in[11]; const float* Bproj = (const float*)d_in[12];
    const float* g2 = (const float*)d_in[13]; const float* b2 = (const float*)d_in[14];
    const float* W1 = (const float*)d_in[15]; const float* bfc1 = (const float*)d_in[16];
    const float* A1 = (const float*)d_in[17]; const float* B1 = (const float*)d_in[18];
    const float* W2 = (const float*)d_in[19]; const float* bfc2 = (const float*)d_in[20];
    const float* A2 = (const float*)d_in[21]; const float* B2 = (const float*)d_in[22];
    float* out = (float*)d_out;

    // bump allocator over workspace
    char* ws = (char*)d_ws;
    size_t off = 0;
    auto alloc = [&](size_t bytes) -> void* {
        off = (off + 255) & ~(size_t)255;
        void* p = ws + off;
        off += bytes;
        return p;
    };
    const size_t N = NTOK;
    bf16_t* wqkvb  = (bf16_t*)alloc((size_t)CDIM * 3 * CDIM * 2);
    bf16_t* aqkvb  = (bf16_t*)alloc((size_t)CDIM * LRANK * 2);
    bf16_t* bqkvbm = (bf16_t*)alloc((size_t)LRANK * 3 * CDIM * 2);
    bf16_t* wprojb = (bf16_t*)alloc((size_t)CDIM * CDIM * 2);
    bf16_t* aprojb = (bf16_t*)alloc((size_t)CDIM * LRANK * 2);
    bf16_t* bprojbm= (bf16_t*)alloc((size_t)LRANK * CDIM * 2);
    bf16_t* w1b    = (bf16_t*)alloc((size_t)CDIM * HIDD * 2);
    bf16_t* a1b    = (bf16_t*)alloc((size_t)CDIM * LRANK * 2);
    bf16_t* b1bm   = (bf16_t*)alloc((size_t)LRANK * HIDD * 2);
    bf16_t* w2b    = (bf16_t*)alloc((size_t)HIDD * CDIM * 2);
    bf16_t* a2b    = (bf16_t*)alloc((size_t)HIDD * LRANK * 2);
    bf16_t* b2bm   = (bf16_t*)alloc((size_t)LRANK * CDIM * 2);
    bf16_t* xln1b  = (bf16_t*)alloc(N * CDIM * 2);
    float*  qkvf   = (float*) alloc(N * 3 * CDIM * 4);
    bf16_t* qkvb   = (bf16_t*)alloc(N * 3 * CDIM * 2);
    float*  t1f    = (float*) alloc(N * LRANK * 4);
    bf16_t* t1b    = (bf16_t*)alloc(N * LRANK * 2);
    bf16_t* obf    = (bf16_t*)alloc(N * CDIM * 2);
    float*  feat1  = (float*) alloc(N * CDIM * 4);
    float*  t2f    = (float*) alloc(N * LRANK * 4);
    bf16_t* t2b    = (bf16_t*)alloc(N * LRANK * 2);
    bf16_t* xln2b  = (bf16_t*)alloc(N * CDIM * 2);
    float*  h0f    = (float*) alloc(N * HIDD * 4);
    bf16_t* hb     = (bf16_t*)alloc(N * HIDD * 2);
    float*  t3f    = (float*) alloc(N * LRANK * 4);
    bf16_t* t3b    = (bf16_t*)alloc(N * LRANK * 2);
    float*  t4f    = (float*) alloc(N * LRANK * 4);
    bf16_t* t4b    = (bf16_t*)alloc(N * LRANK * 2);

    auto cast = [&](const float* s, bf16_t* d, long long n) {
        int blocks = (int)((n + 2047) / 2048);
        if (blocks > 16384) blocks = 16384;
        cast_f32_bf16_kernel<<<blocks, 256, 0, stream>>>(s, d, n);
    };
    auto gemm = [&](const bf16_t* A, const bf16_t* B, const float* bias,
                    const float* res, float* Cf, bf16_t* Cb,
                    int M, int Nc, int Kd, float alpha, int accum) {
        dim3 grid((Nc + BN - 1) / BN, (M + BM - 1) / BM);
        gemm_bf16_kernel<<<grid, 256, 0, stream>>>(A, B, bias, res, Cf, Cb,
                                                   M, Nc, Kd, alpha, accum);
    };

    // weights -> bf16
    cast(Wqkv,  wqkvb,  (long long)CDIM * 3 * CDIM);
    cast(Aqkv,  aqkvb,  (long long)CDIM * LRANK);
    cast(Bqkv,  bqkvbm, (long long)LRANK * 3 * CDIM);
    cast(Wproj, wprojb, (long long)CDIM * CDIM);
    cast(Aproj, aprojb, (long long)CDIM * LRANK);
    cast(Bproj, bprojbm,(long long)LRANK * CDIM);
    cast(W1,    w1b,    (long long)CDIM * HIDD);
    cast(A1,    a1b,    (long long)CDIM * LRANK);
    cast(B1,    b1bm,   (long long)LRANK * HIDD);
    cast(W2,    w2b,    (long long)HIDD * CDIM);
    cast(A2,    a2b,    (long long)HIDD * LRANK);
    cast(B2,    b2bm,   (long long)LRANK * CDIM);

    // --- attention block ---
    ln_cast_kernel<<<NTOK, 256, 0, stream>>>(feat, g1, b1, xln1b);
    gemm(xln1b, wqkvb, bqkv, nullptr, qkvf, nullptr, NTOK, 3 * CDIM, CDIM, 1.0f, 0);
    gemm(xln1b, aqkvb, nullptr, nullptr, t1f, t1b, NTOK, LRANK, CDIM, 1.0f, 0);
    gemm(t1b, bqkvbm, nullptr, nullptr, qkvf, qkvb, NTOK, 3 * CDIM, LRANK, LORA_SCALE, 1);
    attn_kernel<<<dim3(NWIN, NHEAD), 256, 0, stream>>>(qkvb, order, obf);
    gemm(obf, wprojb, bproj, feat, feat1, nullptr, NTOK, CDIM, CDIM, 1.0f, 0);
    gemm(obf, aprojb, nullptr, nullptr, t2f, t2b, NTOK, LRANK, CDIM, 1.0f, 0);
    gemm(t2b, bprojbm, nullptr, nullptr, feat1, nullptr, NTOK, CDIM, LRANK, LORA_SCALE, 1);

    // --- MLP block ---
    ln_cast_kernel<<<NTOK, 256, 0, stream>>>(feat1, g2, b2, xln2b);
    gemm(xln2b, w1b, bfc1, nullptr, h0f, nullptr, NTOK, HIDD, CDIM, 1.0f, 0);
    gemm(xln2b, a1b, nullptr, nullptr, t3f, t3b, NTOK, LRANK, CDIM, 1.0f, 0);
    gemm(t3b, b1bm, nullptr, nullptr, h0f, nullptr, NTOK, HIDD, LRANK, LORA_SCALE, 1);
    {
        long long n = (long long)NTOK * HIDD;
        int blocks = (int)((n + 2047) / 2048);
        if (blocks > 16384) blocks = 16384;
        gelu_cast_kernel<<<blocks, 256, 0, stream>>>(h0f, hb, n);
    }
    gemm(hb, a2b, nullptr, nullptr, t4f, t4b, NTOK, LRANK, HIDD, 1.0f, 0);
    gemm(hb, w2b, bfc2, feat1, out, nullptr, NTOK, CDIM, HIDD, 1.0f, 0);
    gemm(t4b, b2bm, nullptr, nullptr, out, nullptr, NTOK, CDIM, LRANK, LORA_SCALE, 1);
}
